// TransformerModel_91285234909638
// MI455X (gfx1250) — matrix operand order repaired and checked by reading, not compiled, not run on hardware
//
#include <hip/hip_runtime.h>

// ---------------------------------------------------------------------------
// Transformer forward for MI455X (gfx1250, wave32, WMMA).
// All GEMMs run through v_wmma_f32_16x16x32_bf16 (fp32 accumulate).
// GEMM: 128x128 block tile, 8 waves, double-buffered LDS, b128 coalesced
//       staging with packed bf16 conversion, global_prefetch for L2 warm-up.
// Attention: flash-style, 4 independent wave32's per block, barrier-free
//       (per-wave in-order LDS + explicit s_wait_dscnt).
// ---------------------------------------------------------------------------

typedef __attribute__((ext_vector_type(16))) __bf16 v16bf;
typedef __attribute__((ext_vector_type(8)))  float  v8f;

#define D_MODEL 512
#define NHEAD   8
#define NLAYER  6
#define D_FF    2048
#define VOCAB   32000
#define BATCH   2
#define SEQ     2048
#define DK      64          // D_MODEL / NHEAD
#define PAD_ID  1

__device__ __forceinline__ unsigned bf_bits(float f) {
  // round-to-nearest-even fp32 -> bf16 bits (low 16)
  unsigned u = __builtin_bit_cast(unsigned, f);
  return (u + 0x7FFFu + ((u >> 16) & 1u)) >> 16;
}
__device__ __forceinline__ unsigned pack_bf2(float f0, float f1) {
  return bf_bits(f0) | (bf_bits(f1) << 16);
}
__device__ __forceinline__ __bf16 f2bf(float f) {
  unsigned short h = (unsigned short)bf_bits(f);
  return __builtin_bit_cast(__bf16, h);
}

__device__ __forceinline__ v8f wmma_bf16(v16bf a, v16bf b, v8f c) {
  return __builtin_amdgcn_wmma_f32_16x16x32_bf16(
      /*neg_a=*/false, a, /*neg_b=*/false, b,
      /*c_mod=*/(short)0, c, /*reuse_a=*/false, /*reuse_b=*/false);
}

// Per-wave LDS producer->consumer ordering without a workgroup barrier:
// DS ops from one wave complete in order (ISA 7.3); the wait + memory clobber
// keeps the compiler from reordering and drains outstanding DS ops.
#define LDS_WAVE_FENCE() asm volatile("s_wait_dscnt 0" ::: "memory")

// ---------------------------------------------------------------------------
// Embedding + sinusoidal PE:  out[b,s,:] = emb[tok[b,s],:] + pe[s,:]
// ---------------------------------------------------------------------------
__global__ __launch_bounds__(256) void embed_kernel(
    const int* __restrict__ tok, const float* __restrict__ emb,
    const float* __restrict__ pe, float* __restrict__ outp, int total) {
  int i = blockIdx.x * blockDim.x + threadIdx.x;
  if (i >= total) return;
  int row = i / D_MODEL, d = i - row * D_MODEL;
  int s = row % SEQ;
  outp[i] = emb[(size_t)tok[row] * D_MODEL + d] + pe[(size_t)s * D_MODEL + d];
}

// ---------------------------------------------------------------------------
// Row LayerNorm over D_MODEL (one block per row).
// ---------------------------------------------------------------------------
__global__ __launch_bounds__(256) void layernorm_kernel(
    const float* __restrict__ in, const float* __restrict__ g,
    const float* __restrict__ bta, float* __restrict__ outp) {
  const int row = blockIdx.x;
  const float* x = in + (size_t)row * D_MODEL;
  __shared__ float red[256];
  float s = 0.f;
  for (int i = threadIdx.x; i < D_MODEL; i += 256) s += x[i];
  red[threadIdx.x] = s; __syncthreads();
  for (int st = 128; st; st >>= 1) {
    if (threadIdx.x < st) red[threadIdx.x] += red[threadIdx.x + st];
    __syncthreads();
  }
  const float mean = red[0] * (1.0f / D_MODEL); __syncthreads();
  float vs = 0.f;
  for (int i = threadIdx.x; i < D_MODEL; i += 256) {
    float d = x[i] - mean; vs += d * d;
  }
  red[threadIdx.x] = vs; __syncthreads();
  for (int st = 128; st; st >>= 1) {
    if (threadIdx.x < st) red[threadIdx.x] += red[threadIdx.x + st];
    __syncthreads();
  }
  const float rstd = rsqrtf(red[0] * (1.0f / D_MODEL) + 1e-5f);
  for (int i = threadIdx.x; i < D_MODEL; i += 256)
    outp[(size_t)row * D_MODEL + i] = (x[i] - mean) * rstd * g[i] + bta[i];
}

// ---------------------------------------------------------------------------
// bf16 WMMA GEMM:  C[M,N] = A[M,K] @ Bw[K,N] + bias (+ residual) (ReLU?)
// Block = 8 waves (256 thr), tile 128x128, K-step 32, double-buffered LDS.
// Each wave computes 32x64 (2 M-tiles x 4 N-tiles = 8 WMMA per k-step;
// 64 WMMA per block per k-step). Staging uses coalesced global_load_b128
// with fp32 -> bf16 conversion in flight (B transposed into LDS) so every
// WMMA fragment is contiguous 16B LDS reads matching ISA 7.12.2 layouts.
// Requires M%128==N%128==K%32==0.  flags: bit0 = ReLU.
// ---------------------------------------------------------------------------
__global__ __launch_bounds__(256) void gemm_kernel(
    const float* __restrict__ A, const float* __restrict__ Bw,
    const float* __restrict__ bias, const float* res, float* C,
    int M, int N, int K, int flags) {
  __shared__ __align__(16) unsigned AsU[2][128][16];  // bf16x2 : [m][k/2]
  __shared__ __align__(16) unsigned BsU[2][128][16];  // bf16x2 : [n][k/2]
  const int tid  = threadIdx.x;
  const int w    = tid >> 5;
  const int lane = tid & 31;
  const int wr = w >> 1, wc = w & 1;          // 4x2 wave grid
  const int lo = lane & 15, hi = lane >> 4;
  const long row0 = (long)blockIdx.y * 128;
  const long col0 = (long)blockIdx.x * 128;

  v8f acc[2][4] = {};

  auto stage = [&](int buf, int k0) {
    // A tile 128x32: b128 loads (4 k-values), packed b64 LDS stores.
    // 1024 quads / 256 thr = 4 per thread; 8 lanes cover one 128B row chunk.
    for (int i = tid; i < 128 * 8; i += 256) {
      const int r = i >> 3, cq = i & 7;
      const float4 a4 = *(const float4*)&A[(row0 + r) * K + (k0 + cq * 4)];
      *(uint2*)&AsU[buf][r][cq * 2] =
          uint2{pack_bf2(a4.x, a4.y), pack_bf2(a4.z, a4.w)};
    }
    // B tile 32x128: row-major coalesced b128 loads (4 consecutive n),
    // transpose via 4 b16 LDS scatter stores into [n][k] layout.
    __bf16* Bsb = (__bf16*)BsU[buf];
    for (int i = tid; i < 32 * 32; i += 256) {
      const int kk = i >> 5, nq = i & 31;
      const float4 b4 =
          *(const float4*)&Bw[(size_t)(k0 + kk) * N + (col0 + nq * 4)];
      Bsb[(nq * 4 + 0) * 32 + kk] = f2bf(b4.x);
      Bsb[(nq * 4 + 1) * 32 + kk] = f2bf(b4.y);
      Bsb[(nq * 4 + 2) * 32 + kk] = f2bf(b4.z);
      Bsb[(nq * 4 + 3) * 32 + kk] = f2bf(b4.w);
    }
  };

  stage(0, 0);
  const int nsteps = K / 32;
  for (int it = 0; it < nsteps; ++it) {
    const int cur = it & 1;
    __syncthreads();                         // stage(cur) visible to all waves
    if (it + 1 < nsteps) stage(cur ^ 1, (it + 1) * 32);
    if (it + 2 < nsteps) {                   // L2 warm-up two steps ahead
      __builtin_prefetch(&A[(row0 + (tid >> 1)) * K + (it + 2) * 32 + (tid & 1) * 16], 0, 1);
      __builtin_prefetch(&Bw[(size_t)((it + 2) * 32 + lane) * N + col0 + w * 16], 0, 1);
    }
    const __bf16* Asb = (const __bf16*)AsU[cur];
    const __bf16* Bsb = (const __bf16*)BsU[cur];
    v16bf af[2];
#pragma unroll
    for (int mt = 0; mt < 2; ++mt) {
      const int r = wr * 32 + mt * 16 + lo;
      ((float4*)&af[mt])[0] = *(const float4*)&Asb[r * 32 + hi * 8];
      ((float4*)&af[mt])[1] = *(const float4*)&Asb[r * 32 + 16 + hi * 8];
    }
#pragma unroll
    for (int nt = 0; nt < 4; ++nt) {
      const int n = wc * 64 + nt * 16 + lo;
      v16bf bf_;
      ((float4*)&bf_)[0] = *(const float4*)&Bsb[n * 32 + hi * 8];
      ((float4*)&bf_)[1] = *(const float4*)&Bsb[n * 32 + 16 + hi * 8];
#pragma unroll
      for (int mt = 0; mt < 2; ++mt)
        acc[mt][nt] = wmma_bf16(af[mt], bf_, acc[mt][nt]);
    }
  }
  // Epilogue: C-layout element r -> row = r + 8*hi, col = lo (within tile)
#pragma unroll
  for (int mt = 0; mt < 2; ++mt)
#pragma unroll
    for (int nt = 0; nt < 4; ++nt) {
      const long col = col0 + wc * 64 + nt * 16 + lo;
      const float bz = bias ? bias[col] : 0.f;
#pragma unroll
      for (int r = 0; r < 8; ++r) {
        const long row = row0 + wr * 32 + mt * 16 + hi * 8 + r;
        float val = acc[mt][nt][r] + bz;
        if (flags & 1) val = fmaxf(val, 0.f);
        if (res) val += res[row * N + col];
        C[row * N + col] = val;
      }
    }
}

// ---------------------------------------------------------------------------
// Flash attention. 4 independent wave32's per block; wave w handles query
// tile (blockIdx.x*4 + w) of 16 queries for (batch, head). Each wave owns a
// private LDS slice, so no workgroup barriers are needed (causal waves have
// divergent chunk counts). Mask from key tokens (PAD=1) + optional causal.
// ---------------------------------------------------------------------------
__global__ __launch_bounds__(128) void flash_attn_kernel(
    const float* __restrict__ Q, const float* __restrict__ Kp,
    const float* __restrict__ Vp, const int* __restrict__ keytok,
    float* __restrict__ O, int causal) {
  const int w = threadIdx.x >> 5, lane = threadIdx.x & 31;
  const int qt = blockIdx.x * 4 + w;
  const int h = blockIdx.y, b = blockIdx.z;
  const int lo = lane & 15, hi = lane >> 4;
  const int Dm = NHEAD * DK;

  __shared__ __align__(16) __bf16 Ps[4][16][32];   // per-wave P tile
  __shared__ __align__(16) __bf16 Vs[4][64][32];   // per-wave V^T chunk

  // Q tile (16 x 64), pre-scaled by 1/sqrt(dk)=0.125, as two A fragments.
  v16bf qf[2];
  {
    const float* qrow = Q + ((size_t)(b * SEQ + qt * 16 + lo)) * Dm + h * DK;
#pragma unroll
    for (int st = 0; st < 2; ++st)
#pragma unroll
      for (int e = 0; e < 16; ++e)
        qf[st][e] = f2bf(qrow[st * 32 + (e >> 3) * 16 + hi * 8 + (e & 7)] * 0.125f);
  }

  float mrow[8], lrow[8];
#pragma unroll
  for (int r = 0; r < 8; ++r) { mrow[r] = -1e30f; lrow[r] = 0.f; }
  v8f o[4] = {};

  const int nchunks = causal ? ((qt * 16 + 16 + 31) / 32) : (SEQ / 32);
  for (int c = 0; c < nchunks; ++c) {
    float p[2][8];
#pragma unroll
    for (int t = 0; t < 2; ++t) {
      v8f s = {};
      const int kt = c * 2 + t;
      // B fragment of K^T: col = key (kt*16+lo), K-dim = feature d (contig)
      const float* krow = Kp + ((size_t)(b * SEQ + kt * 16 + lo)) * Dm + h * DK;
#pragma unroll
      for (int st = 0; st < 2; ++st) {
        v16bf kf;
#pragma unroll
        for (int e = 0; e < 16; ++e) kf[e] = f2bf(krow[st * 32 + (e >> 3) * 16 + hi * 8 + (e & 7)]);
        s = wmma_bf16(qf[st], kf, s);
      }
      const int key = kt * 16 + lo;
      const int tok = keytok[b * SEQ + key];
#pragma unroll
      for (int r = 0; r < 8; ++r) {
        const int qg = qt * 16 + hi * 8 + r;
        const bool ok = (tok != PAD_ID) && (!causal || key <= qg);
        p[t][r] = ok ? s[r] : -1e30f;
      }
    }
    // Online softmax: each row lives across 16 lanes of a half-wave.
#pragma unroll
    for (int r = 0; r < 8; ++r) {
      float mx = fmaxf(p[0][r], p[1][r]);
#pragma unroll
      for (int off = 8; off; off >>= 1) mx = fmaxf(mx, __shfl_xor(mx, off, 32));
      const float nm = fmaxf(mrow[r], mx);
      const float alpha = __expf(mrow[r] - nm);
      mrow[r] = nm;
      p[0][r] = __expf(p[0][r] - nm);
      p[1][r] = __expf(p[1][r] - nm);
      float rs = p[0][r] + p[1][r];
#pragma unroll
      for (int off = 8; off; off >>= 1) rs += __shfl_xor(rs, off, 32);
      lrow[r] = lrow[r] * alpha + rs;
#pragma unroll
      for (int nt = 0; nt < 4; ++nt) o[nt][r] *= alpha;
    }
    // Re-layout P (C-layout -> A-layout) and stage V^T through this wave's
    // private LDS slice; per-wave DS ordering is in-order.
#pragma unroll
    for (int t = 0; t < 2; ++t)
#pragma unroll
      for (int r = 0; r < 8; ++r)
        Ps[w][hi * 8 + r][t * 16 + lo] = f2bf(p[t][r]);
    for (int i = lane; i < 32 * 64; i += 32) {
      const int kk = i >> 6, nn = i & 63;
      Vs[w][nn][kk] = f2bf(Vp[((size_t)(b * SEQ + c * 32 + kk)) * Dm + h * DK + nn]);
    }
    LDS_WAVE_FENCE();
    v16bf pf;
    ((float4*)&pf)[0] = *(const float4*)&Ps[w][lo][hi * 8];
    ((float4*)&pf)[1] = *(const float4*)&Ps[w][lo][16 + hi * 8];
#pragma unroll
    for (int nt = 0; nt < 4; ++nt) {
      v16bf vf;
      ((float4*)&vf)[0] = *(const float4*)&Vs[w][nt * 16 + lo][hi * 8];
      ((float4*)&vf)[1] = *(const float4*)&Vs[w][nt * 16 + lo][16 + hi * 8];
      o[nt] = wmma_bf16(pf, vf, o[nt]);
    }
    LDS_WAVE_FENCE();
  }
  float* obase = O + ((size_t)(b * SEQ + qt * 16)) * Dm + h * DK;
#pragma unroll
  for (int nt = 0; nt < 4; ++nt)
#pragma unroll
    for (int r = 0; r < 8; ++r)
      obase[(size_t)(hi * 8 + r) * Dm + nt * 16 + lo] = o[nt][r] / lrow[r];
}

// ---------------------------------------------------------------------------
// Host orchestration
// ---------------------------------------------------------------------------
static inline void run_gemm(const float* A, const float* Bw, const float* bias,
                            const float* res, float* C, int M, int N, int K,
                            int flags, hipStream_t s) {
  gemm_kernel<<<dim3(N / 128, M / 128), dim3(256), 0, s>>>(A, Bw, bias, res, C,
                                                           M, N, K, flags);
}

extern "C" void kernel_launch(void* const* d_in, const int* in_sizes, int n_in,
                              void* d_out, int out_size, void* d_ws,
                              size_t ws_size, hipStream_t stream) {
  (void)in_sizes; (void)n_in; (void)out_size; (void)ws_size;
  const int M  = BATCH * SEQ;          // 4096
  const size_t DD = (size_t)D_MODEL * D_MODEL;

  const int*   src       = (const int*)  d_in[0];
  const int*   tgt       = (const int*)  d_in[1];
  const float* src_emb   = (const float*)d_in[2];
  const float* tgt_emb   = (const float*)d_in[3];
  const float* pe        = (const float*)d_in[4];
  const float* enc_attn_w = (const float*)d_in[5];
  const float* enc_attn_b = (const float*)d_in[6];
  const float* enc_ff_w1  = (const float*)d_in[7];
  const float* enc_ff_b1  = (const float*)d_in[8];
  const float* enc_ff_w2  = (const float*)d_in[9];
  const float* enc_ff_b2  = (const float*)d_in[10];
  const float* enc_ln_g   = (const float*)d_in[11];
  const float* enc_ln_b   = (const float*)d_in[12];
  const float* dec_self_w = (const float*)d_in[13];
  const float* dec_self_b = (const float*)d_in[14];
  const float* dec_src_w  = (const float*)d_in[15];
  const float* dec_src_b  = (const float*)d_in[16];
  const float* dec_ff_w1  = (const float*)d_in[17];
  const float* dec_ff_b1  = (const float*)d_in[18];
  const float* dec_ff_w2  = (const float*)d_in[19];
  const float* dec_ff_b2  = (const float*)d_in[20];
  const float* dec_ln_g   = (const float*)d_in[21];
  const float* dec_ln_b   = (const float*)d_in[22];
  const float* ln_g       = (const float*)d_in[23];
  const float* ln_b       = (const float*)d_in[24];
  const float* fc_w       = (const float*)d_in[25];
  const float* fc_b       = (const float*)d_in[26];
  float* out = (float*)d_out;

  // Workspace layout (fp32). Total = 3*M*D + M*DFF floats ~= 60 MB.
  float* x   = (float*)d_ws;                      // enc stream -> memory
  float* y   = x  + (size_t)M * D_MODEL;          // dec stream
  float* x2  = y  + (size_t)M * D_MODEL;          // LN output
  float* qb  = x2 + (size_t)M * D_MODEL;          // q
  float* kb  = qb + (size_t)M * D_MODEL;          // k
  float* vb  = kb + (size_t)M * D_MODEL;          // v
  float* ab  = vb + (size_t)M * D_MODEL;          // attention output
  float* ffh = qb;   // FF hidden [M,DFF] aliases qb..ab (exactly 4*M*D floats)

  const int total = M * D_MODEL;
  const dim3 attn_grid(SEQ / (16 * 4), NHEAD, BATCH);

  // ---------------- Encoder ----------------
  embed_kernel<<<(total + 255) / 256, 256, 0, stream>>>(src, src_emb, pe, x, total);
  for (int l = 0; l < NLAYER; ++l) {
    const float* W  = enc_attn_w + (size_t)l * 4 * DD;
    const float* Wb = enc_attn_b + (size_t)l * 4 * D_MODEL;
    layernorm_kernel<<<M, 256, 0, stream>>>(
        x, enc_ln_g + (size_t)(l * 2 + 0) * D_MODEL,
        enc_ln_b + (size_t)(l * 2 + 0) * D_MODEL, x2);
    run_gemm(x2, W + 0 * DD, Wb + 0 * D_MODEL, nullptr, qb, M, D_MODEL, D_MODEL, 0, stream);
    run_gemm(x2, W + 1 * DD, Wb + 1 * D_MODEL, nullptr, kb, M, D_MODEL, D_MODEL, 0, stream);
    run_gemm(x2, W + 2 * DD, Wb + 2 * D_MODEL, nullptr, vb, M, D_MODEL, D_MODEL, 0, stream);
    flash_attn_kernel<<<attn_grid, 128, 0, stream>>>(qb, kb, vb, src, ab, 0);
    run_gemm(ab, W + 3 * DD, Wb + 3 * D_MODEL, x, x, M, D_MODEL, D_MODEL, 0, stream);
    layernorm_kernel<<<M, 256, 0, stream>>>(
        x, enc_ln_g + (size_t)(l * 2 + 1) * D_MODEL,
        enc_ln_b + (size_t)(l * 2 + 1) * D_MODEL, x2);
    run_gemm(x2, enc_ff_w1 + (size_t)l * D_MODEL * D_FF,
             enc_ff_b1 + (size_t)l * D_FF, nullptr, ffh, M, D_FF, D_MODEL, 1, stream);
    run_gemm(ffh, enc_ff_w2 + (size_t)l * D_FF * D_MODEL,
             enc_ff_b2 + (size_t)l * D_MODEL, x, x, M, D_MODEL, D_FF, 0, stream);
  }
  // x now holds encoder memory.

  // ---------------- Decoder ----------------
  embed_kernel<<<(total + 255) / 256, 256, 0, stream>>>(tgt, tgt_emb, pe, y, total);
  for (int l = 0; l < NLAYER; ++l) {
    const float* Ws = dec_self_w + (size_t)l * 4 * DD;
    const float* bs = dec_self_b + (size_t)l * 4 * D_MODEL;
    const float* Wc = dec_src_w + (size_t)l * 4 * DD;
    const float* bc = dec_src_b + (size_t)l * 4 * D_MODEL;
    // masked self-attention
    layernorm_kernel<<<M, 256, 0, stream>>>(
        y, dec_ln_g + (size_t)(l * 3 + 0) * D_MODEL,
        dec_ln_b + (size_t)(l * 3 + 0) * D_MODEL, x2);
    run_gemm(x2, Ws + 0 * DD, bs + 0 * D_MODEL, nullptr, qb, M, D_MODEL, D_MODEL, 0, stream);
    run_gemm(x2, Ws + 1 * DD, bs + 1 * D_MODEL, nullptr, kb, M, D_MODEL, D_MODEL, 0, stream);
    run_gemm(x2, Ws + 2 * DD, bs + 2 * D_MODEL, nullptr, vb, M, D_MODEL, D_MODEL, 0, stream);
    flash_attn_kernel<<<attn_grid, 128, 0, stream>>>(qb, kb, vb, tgt, ab, 1);
    run_gemm(ab, Ws + 3 * DD, bs + 3 * D_MODEL, y, y, M, D_MODEL, D_MODEL, 0, stream);
    // cross-attention (q from y2, k/v from encoder memory)
    layernorm_kernel<<<M, 256, 0, stream>>>(
        y, dec_ln_g + (size_t)(l * 3 + 1) * D_MODEL,
        dec_ln_b + (size_t)(l * 3 + 1) * D_MODEL, x2);
    run_gemm(x2, Wc + 0 * DD, bc + 0 * D_MODEL, nullptr, qb, M, D_MODEL, D_MODEL, 0, stream);
    run_gemm(x,  Wc + 1 * DD, bc + 1 * D_MODEL, nullptr, kb, M, D_MODEL, D_MODEL, 0, stream);
    run_gemm(x,  Wc + 2 * DD, bc + 2 * D_MODEL, nullptr, vb, M, D_MODEL, D_MODEL, 0, stream);
    flash_attn_kernel<<<attn_grid, 128, 0, stream>>>(qb, kb, vb, src, ab, 0);
    run_gemm(ab, Wc + 3 * DD, bc + 3 * D_MODEL, y, y, M, D_MODEL, D_MODEL, 0, stream);
    // feed-forward
    layernorm_kernel<<<M, 256, 0, stream>>>(
        y, dec_ln_g + (size_t)(l * 3 + 2) * D_MODEL,
        dec_ln_b + (size_t)(l * 3 + 2) * D_MODEL, x2);
    run_gemm(x2, dec_ff_w1 + (size_t)l * D_MODEL * D_FF,
             dec_ff_b1 + (size_t)l * D_FF, nullptr, ffh, M, D_FF, D_MODEL, 1, stream);
    run_gemm(ffh, dec_ff_w2 + (size_t)l * D_FF * D_MODEL,
             dec_ff_b2 + (size_t)l * D_MODEL, y, y, M, D_MODEL, D_FF, 0, stream);
  }

  // ---------------- Final LN + vocab projection ----------------
  layernorm_kernel<<<M, 256, 0, stream>>>(y, ln_g, ln_b, x2);
  run_gemm(x2, fc_w, fc_b, nullptr, out, M, VOCAB, D_MODEL, 0, stream);
}
